// get_model_40321152975317
// MI455X (gfx1250) — compile-verified, matrix operand order checked by reading, not run on hardware
//
#include <hip/hip_runtime.h>
#include <cstdint>
#include <cstddef>

// ---------------------------------------------------------------------------
// PointNet++ MSG (B=8, N=4096, CIN=7, NOUT=3) for gfx1250 (MI455X).
// All MLP einsums run through V_WMMA_F32_16X16X32_F16 (f32 accum, f16 data).
// Activations live in K-padded (mult of 32) f16 ping/pong buffers so every
// WMMA fragment load is one 16-byte global_load_b128 per half-fragment.
// FPS caches the point cloud + running min-distance in LDS (<=64KB per WG).
// ---------------------------------------------------------------------------

typedef __attribute__((ext_vector_type(8)))  _Float16 v8h;
typedef __attribute__((ext_vector_type(16))) _Float16 v16h;
typedef __attribute__((ext_vector_type(8)))  float    v8f;
typedef _Float16 f16;

#define BATCH 8
#define NPTS  4096

static inline int r32i(int x) { return (x + 31) & ~31; }

// -------------------------------- kernels ----------------------------------

// xyz (B,7,N) -> p (B,N,7) and l0_xyz (B,N,3)
__global__ void k_transpose(const float* __restrict__ x, float* __restrict__ p,
                            float* __restrict__ xyz0, int C, int N)
{
  int t = blockIdx.x * blockDim.x + threadIdx.x;
  int total = BATCH * N * C;
  if (t >= total) return;
  int c = t % C;
  int i = (t / C) % N;
  int b = t / (C * N);
  float v = x[((size_t)b * C + c) * N + i];
  p[t] = v;
  if (c < 3) xyz0[((size_t)b * N + i) * 3 + c] = v;
}

// Farthest point sampling: one workgroup per batch. Writes new_xyz directly.
__global__ void k_fps(const float* __restrict__ xyz, int n, int S,
                      float* __restrict__ new_xyz)
{
  extern __shared__ float sm[];
  float* sx = sm;
  float* sy = sm + n;
  float* sz = sm + 2 * n;
  float* sd = sm + 3 * n;
  __shared__ float rv[256];
  __shared__ int   ri[256];
  __shared__ int   s_far;

  int b = blockIdx.x;
  int tid = threadIdx.x;
  for (int i = tid; i < n; i += blockDim.x) {
    size_t q = ((size_t)b * n + i) * 3;
    sx[i] = xyz[q + 0];
    sy[i] = xyz[q + 1];
    sz[i] = xyz[q + 2];
    sd[i] = 1e10f;
  }
  if (tid == 0) s_far = 0;
  __syncthreads();

  for (int it = 0; it < S; it++) {
    int far = s_far;
    if (tid == 0) {
      size_t q = ((size_t)b * S + it) * 3;
      new_xyz[q + 0] = sx[far];
      new_xyz[q + 1] = sy[far];
      new_xyz[q + 2] = sz[far];
    }
    float cx = sx[far], cy = sy[far], cz = sz[far];
    float best = -1.0f;
    int bi = 0x7fffffff;
    for (int i = tid; i < n; i += blockDim.x) {
      float dx = sx[i] - cx, dy = sy[i] - cy, dz = sz[i] - cz;
      float d = dx * dx + dy * dy + dz * dz;
      float dd = sd[i];
      dd = dd < d ? dd : d;
      sd[i] = dd;
      if (dd > best) { best = dd; bi = i; }   // strict > keeps lowest index
    }
    rv[tid] = best;
    ri[tid] = bi;
    __syncthreads();
    for (int s = 128; s > 0; s >>= 1) {
      if (tid < s) {
        float ov = rv[tid + s];
        int   oi = ri[tid + s];
        if (ov > rv[tid] || (ov == rv[tid] && oi < ri[tid])) {
          rv[tid] = ov; ri[tid] = oi;
        }
      }
      __syncthreads();
    }
    if (tid == 0) s_far = ri[0];
    __syncthreads();
  }
}

// Ball query: first k indices (ascending) with d^2 <= r^2, padded with first.
__global__ void k_ballquery(const float* __restrict__ xyz,
                            const float* __restrict__ nxyz,
                            int* __restrict__ gidx,
                            int n, int S, int k, float r2)
{
  int t = blockIdx.x * blockDim.x + threadIdx.x;
  if (t >= BATCH * S) return;
  int b = t / S, s = t % S;
  size_t cq = ((size_t)b * S + s) * 3;
  float cx = nxyz[cq + 0], cy = nxyz[cq + 1], cz = nxyz[cq + 2];
  int* g = gidx + (size_t)t * k;
  int cnt = 0, first = 0;
  for (int j = 0; j < n && cnt < k; j++) {
    size_t q = ((size_t)b * n + j) * 3;
    float dx = xyz[q + 0] - cx, dy = xyz[q + 1] - cy, dz = xyz[q + 2] - cz;
    float d = dx * dx + dy * dy + dz * dz;
    if (d <= r2) { if (cnt == 0) first = j; g[cnt++] = j; }
  }
  for (; cnt < k; cnt++) g[cnt] = first;
}

// Gather grouped features: out[(b,s,kk), 0:C]=points, [C:C+3]=xyz-center, pad 0
__global__ void k_group(const float* __restrict__ pts, const float* __restrict__ xyz,
                        const float* __restrict__ nxyz, const int* __restrict__ gidx,
                        f16* __restrict__ out, int n, int S, int k, int C, int Kp)
{
  size_t t = (size_t)blockIdx.x * blockDim.x + threadIdx.x;
  size_t total = (size_t)BATCH * S * k * Kp;
  if (t >= total) return;
  int col = (int)(t % Kp);
  size_t row = t / Kp;
  int kk = (int)(row % k);
  int s = (int)((row / k) % S);
  int b = (int)(row / ((size_t)k * S));
  (void)kk;
  int j = gidx[row];
  float v = 0.f;
  if (col < C) {
    v = pts[((size_t)b * n + j) * C + col];
  } else if (col < C + 3) {
    int d = col - C;
    v = xyz[((size_t)b * n + j) * 3 + d] - nxyz[((size_t)b * S + s) * 3 + d];
  }
  out[t] = (f16)v;
}

// Weight f32(O,C) -> f16 padded (Op,Kp), zero-filled.
__global__ void k_wconv(const float* __restrict__ w, f16* __restrict__ wh,
                        int O, int C, int Op, int Kp)
{
  size_t t = (size_t)blockIdx.x * blockDim.x + threadIdx.x;
  size_t total = (size_t)Op * Kp;
  if (t >= total) return;
  int kk = (int)(t % Kp);
  int o  = (int)(t / Kp);
  wh[t] = (o < O && kk < C) ? (f16)w[(size_t)o * C + kk] : (f16)0.f;
}

// WMMA GEMM: out[m,o] = act( sum_c A[m,c]*W[o,c] + bias[o] ).
// One wave per 16x16 tile; EXEC all-ones around the WMMA (block=32).
// A: (M,Kp) f16 padded. W: (Op,Kp) f16 padded. outh: (M,Op) f16, outf: (M,O) f32.
__global__ void __launch_bounds__(32)
k_gemm(const f16* __restrict__ A, const f16* __restrict__ W,
       const float* __restrict__ bias,
       f16* __restrict__ outh, float* __restrict__ outf,
       int M, int O, int Kp, int Op, int relu)
{
  int lane = threadIdx.x;
  int hi = lane >> 4;
  int lr = lane & 15;
  int m0 = blockIdx.x * 16;
  int o0 = blockIdx.y * 16;

  const f16* ap = A + (size_t)(m0 + lr) * Kp + hi * 8;
  const f16* wp = W + (size_t)(o0 + lr) * Kp + hi * 8;

  v8f acc = {};
  for (int kt = 0; kt < Kp; kt += 32) {
    // A 16x32 f16 fragment: halves 0..7 -> K = kt+hi*8+0..7,
    //                       halves 8..15 -> K = kt+16+hi*8+0..7   (ISA 7.12.2)
    v8h a0 = *(const v8h*)(ap + kt);
    v8h a1 = *(const v8h*)(ap + kt + 16);
    v8h b0 = *(const v8h*)(wp + kt);
    v8h b1 = *(const v8h*)(wp + kt + 16);
    v16h av = __builtin_shufflevector(a0, a1, 0,1,2,3,4,5,6,7,8,9,10,11,12,13,14,15);
    v16h bv = __builtin_shufflevector(b0, b1, 0,1,2,3,4,5,6,7,8,9,10,11,12,13,14,15);
    acc = __builtin_amdgcn_wmma_f32_16x16x32_f16(
        /*neg_a=*/false, av, /*neg_b=*/false, bv,
        /*c_mod=*/(short)0, acc, /*reuse_a=*/false, /*reuse_b=*/false);
  }

  int o = o0 + lr;
#pragma unroll
  for (int r = 0; r < 8; r++) {
    int m = m0 + hi * 8 + r;        // C/D layout: VGPR r -> M = r + 8*hi, N = lr
    float v = acc[r];
    if (o < O) v += bias[o]; else v = 0.f;
    if (relu) v = v > 0.f ? v : 0.f;
    if (m < M) {
      if (outh) outh[(size_t)m * Op + o] = (f16)v;
      if (outf && o < O) outf[(size_t)m * O + o] = v;
    }
  }
}

// Max over k samples, scatter into concat channel slot of persistent features.
__global__ void k_maxpool(const f16* __restrict__ in, float* __restrict__ out,
                          int S, int k, int O, int Op, int Ctot, int coff)
{
  int t = blockIdx.x * blockDim.x + threadIdx.x;
  int total = BATCH * S * O;
  if (t >= total) return;
  int o = t % O;
  int s = (t / O) % S;
  int b = t / (O * S);
  const f16* base = in + ((size_t)(b * S + s) * k) * Op + o;
  float mx = -1e30f;
  for (int kk = 0; kk < k; kk++) {
    float v = (float)base[(size_t)kk * Op];
    mx = v > mx ? v : mx;
  }
  out[((size_t)b * S + s) * Ctot + coff + o] = mx;
}

// 3-NN (squared distance) with stable tie-breaking; normalized inv-dist weights.
__global__ void k_knn3(const float* __restrict__ x1, const float* __restrict__ x2,
                       int n1, int n2, int* __restrict__ idx3, float* __restrict__ w3)
{
  int t = blockIdx.x * blockDim.x + threadIdx.x;
  if (t >= BATCH * n1) return;
  int b = t / n1, i = t % n1;
  size_t q = ((size_t)b * n1 + i) * 3;
  float ax = x1[q + 0], ay = x1[q + 1], az = x1[q + 2];
  float d0 = 1e30f, d1 = 1e30f, d2 = 1e30f;
  int i0 = 0, i1 = 0, i2 = 0;
  for (int j = 0; j < n2; j++) {
    size_t p = ((size_t)b * n2 + j) * 3;
    float dx = ax - x2[p + 0], dy = ay - x2[p + 1], dz = az - x2[p + 2];
    float d = dx * dx + dy * dy + dz * dz;
    if (d < d0)      { d2 = d1; i2 = i1; d1 = d0; i1 = i0; d0 = d; i0 = j; }
    else if (d < d1) { d2 = d1; i2 = i1; d1 = d;  i1 = j; }
    else if (d < d2) { d2 = d;  i2 = j; }
  }
  float w0 = 1.f / (d0 + 1e-8f), w1 = 1.f / (d1 + 1e-8f), w2 = 1.f / (d2 + 1e-8f);
  float s = w0 + w1 + w2;
  w0 /= s; w1 /= s; w2 /= s;
  idx3[(size_t)t * 3 + 0] = i0; idx3[(size_t)t * 3 + 1] = i1; idx3[(size_t)t * 3 + 2] = i2;
  w3[(size_t)t * 3 + 0] = w0;  w3[(size_t)t * 3 + 1] = w1;  w3[(size_t)t * 3 + 2] = w2;
}

// out[(b,i), 0:C1]=points1, [C1:C1+C2]=sum_j w_j * points2[idx_j], pad 0 (f16)
__global__ void k_interpcat(const float* __restrict__ p1, const float* __restrict__ p2,
                            const int* __restrict__ idx3, const float* __restrict__ w3,
                            f16* __restrict__ out, int n1, int n2, int C1, int C2, int Kp)
{
  size_t t = (size_t)blockIdx.x * blockDim.x + threadIdx.x;
  size_t total = (size_t)BATCH * n1 * Kp;
  if (t >= total) return;
  int col = (int)(t % Kp);
  size_t row = t / Kp;
  int b = (int)(row / n1);
  int i = (int)(row % n1);
  float v = 0.f;
  if (col < C1) {
    v = p1[((size_t)b * n1 + i) * C1 + col];
  } else if (col < C1 + C2) {
    int c = col - C1;
    size_t q = ((size_t)b * n1 + i) * 3;
    const float* pb = p2 + (size_t)b * n2 * C2;
    v = w3[q + 0] * pb[(size_t)idx3[q + 0] * C2 + c]
      + w3[q + 1] * pb[(size_t)idx3[q + 1] * C2 + c]
      + w3[q + 2] * pb[(size_t)idx3[q + 2] * C2 + c];
  }
  out[t] = (f16)v;
}

// ------------------------------- host side ---------------------------------

extern "C" void kernel_launch(void* const* d_in, const int* in_sizes, int n_in,
                              void* d_out, int out_size, void* d_ws, size_t ws_size,
                              hipStream_t stream)
{
  (void)in_sizes; (void)n_in; (void)out_size; (void)ws_size;
  const int B = BATCH, N = NPTS;

  static const int SA[4][2][3][2] = {
    { {{16,10},{16,16},{32,16}},   {{32,10},{32,32},{64,32}} },
    { {{64,99},{64,64},{128,64}},  {{64,99},{96,64},{128,96}} },
    { {{128,259},{196,128},{256,196}}, {{128,259},{196,128},{256,196}} },
    { {{256,515},{256,256},{512,256}}, {{256,515},{384,256},{512,384}} },
  };
  static const int NPT[4] = {1024, 256, 64, 16};
  static const float RAD[4][2] = {{0.05f,0.1f},{0.1f,0.2f},{0.2f,0.4f},{0.4f,0.8f}};
  static const int NSAMP[2] = {16, 32};
  static const int FPSPEC[4][3][2] = {
    {{256,1536},{256,256},{0,0}},
    {{256,512},{256,256},{0,0}},
    {{256,352},{128,256},{0,0}},
    {{128,128},{128,128},{128,128}},
  };
  static const int FPN[4] = {2, 2, 2, 3};
  static const int HEADSPEC[2][2] = {{128,128},{3,128}};

  // --- inputs (insertion-order flattening: xyz, sa[L][br][j](w,b), fp, head)
  const float* xin = (const float*)d_in[0];
  int wi = 1;
  const float *saW[4][2][3], *saB[4][2][3];
  for (int L = 0; L < 4; L++)
    for (int br = 0; br < 2; br++)
      for (int j = 0; j < 3; j++) {
        saW[L][br][j] = (const float*)d_in[wi++];
        saB[L][br][j] = (const float*)d_in[wi++];
      }
  const float *fpW[4][3] = {}, *fpB[4][3] = {};
  for (int t = 0; t < 4; t++)
    for (int j = 0; j < FPN[t]; j++) {
      fpW[t][j] = (const float*)d_in[wi++];
      fpB[t][j] = (const float*)d_in[wi++];
    }
  const float *hW[2], *hB[2];
  for (int j = 0; j < 2; j++) {
    hW[j] = (const float*)d_in[wi++];
    hB[j] = (const float*)d_in[wi++];
  }

  // --- workspace carving
  uintptr_t base = (uintptr_t)d_ws;
  size_t off = 0;
  auto alloc = [&](size_t bytes) -> void* {
    off = (off + 255) & ~(size_t)255;
    void* r = (void*)(base + off);
    off += bytes;
    return r;
  };

  float* pbuf = (float*)alloc((size_t)B * N * 7 * 4);
  float* xyz0 = (float*)alloc((size_t)B * N * 3 * 4);
  float* nx[4];
  for (int L = 0; L < 4; L++) nx[L] = (float*)alloc((size_t)B * NPT[L] * 3 * 4);
  int Ctot[4];
  float* lp[4];
  for (int L = 0; L < 4; L++) {
    Ctot[L] = SA[L][0][2][0] + SA[L][1][2][0];  // 96,256,512,1024
    lp[L] = (float*)alloc((size_t)B * NPT[L] * Ctot[L] * 4);
  }
  float* fpo[4];  // new features for levels 1..3 (index by level)
  fpo[3] = (float*)alloc((size_t)B * 64   * 256 * 4);
  fpo[2] = (float*)alloc((size_t)B * 256  * 256 * 4);
  fpo[1] = (float*)alloc((size_t)B * 1024 * 128 * 4);
  int*   gidx  = (int*)alloc((size_t)B * 1024 * 32 * 4);
  int*   knn_i = (int*)alloc((size_t)B * N * 3 * 4);
  float* knn_w = (float*)alloc((size_t)B * N * 3 * 4);
  const size_t MAXE = (size_t)262144 * 64;  // largest padded activation tile
  f16* ping = (f16*)alloc(MAXE * 2);
  f16* pong = (f16*)alloc(MAXE * 2);

  auto blocks = [](size_t total, int bs) { return dim3((unsigned)((total + bs - 1) / bs)); };

  // --- l0: transpose
  k_transpose<<<blocks((size_t)B * N * 7, 256), 256, 0, stream>>>(xin, pbuf, xyz0, 7, N);

  // --- set abstraction levels
  const float* cxyz = xyz0;
  const float* cpts = pbuf;
  int n = N, Cprev = 7;
  for (int L = 0; L < 4; L++) {
    int S = NPT[L];
    size_t shmem = (size_t)4 * n * sizeof(float);
    k_fps<<<dim3(B), dim3(256), shmem, stream>>>(cxyz, n, S, nx[L]);

    for (int br = 0; br < 2; br++) {
      int k = NSAMP[br];
      float r2 = RAD[L][br] * RAD[L][br];
      k_ballquery<<<blocks((size_t)B * S, 256), 256, 0, stream>>>(cxyz, nx[L], gidx, n, S, k, r2);

      int Cin = Cprev + 3;
      int Kp = r32i(Cin);
      size_t M = (size_t)B * S * k;
      k_group<<<blocks(M * Kp, 256), 256, 0, stream>>>(cpts, cxyz, nx[L], gidx, ping, n, S, k, Cprev, Kp);

      f16* src = ping;
      for (int j = 0; j < 3; j++) {
        int O = SA[L][br][j][0], Cw = SA[L][br][j][1];
        int Kw = r32i(Cw), Op = r32i(O);
        f16* wh = (f16*)alloc((size_t)Op * Kw * 2);
        k_wconv<<<blocks((size_t)Op * Kw, 256), 256, 0, stream>>>(saW[L][br][j], wh, O, Cw, Op, Kw);
        f16* dst = (src == ping) ? pong : ping;
        dim3 gg((unsigned)(M / 16), (unsigned)(Op / 16));
        k_gemm<<<gg, 32, 0, stream>>>(src, wh, saB[L][br][j], dst, (float*)nullptr,
                                      (int)M, O, Kw, Op, 1);
        src = dst;
      }
      int Olast = SA[L][br][2][0], Opl = r32i(Olast);
      int coff = (br == 0) ? 0 : SA[L][0][2][0];
      k_maxpool<<<blocks((size_t)B * S * Olast, 256), 256, 0, stream>>>(
          src, lp[L], S, k, Olast, Opl, Ctot[L], coff);
    }
    cxyz = nx[L];
    cpts = lp[L];
    Cprev = Ctot[L];
    n = S;
  }

  // --- feature propagation levels
  const float* xs[5] = {xyz0, nx[0], nx[1], nx[2], nx[3]};
  const float* fs[5] = {nullptr, lp[0], lp[1], lp[2], lp[3]};
  int Cf[5] = {0, Ctot[0], Ctot[1], Ctot[2], Ctot[3]};
  int np[5] = {N, NPT[0], NPT[1], NPT[2], NPT[3]};
  f16* headsrc = nullptr;

  for (int t = 0; t < 4; t++) {
    int lvl = 3 - t;  // 3,2,1,0
    int n1 = np[lvl], n2 = np[lvl + 1];
    const float* x1 = xs[lvl];
    const float* x2 = xs[lvl + 1];
    const float* p1 = (lvl == 0) ? nullptr : fs[lvl];
    int C1 = (lvl == 0) ? 0 : Cf[lvl];
    const float* p2 = fs[lvl + 1];
    int C2 = Cf[lvl + 1];
    int Cin = C1 + C2, Kp = r32i(Cin);
    size_t M = (size_t)B * n1;

    k_knn3<<<blocks(M, 256), 256, 0, stream>>>(x1, x2, n1, n2, knn_i, knn_w);
    k_interpcat<<<blocks(M * Kp, 256), 256, 0, stream>>>(p1, p2, knn_i, knn_w, ping,
                                                         n1, n2, C1, C2, Kp);
    f16* src = ping;
    for (int j = 0; j < FPN[t]; j++) {
      int O = FPSPEC[t][j][0], Cw = FPSPEC[t][j][1];
      int Kw = r32i(Cw), Op = r32i(O);
      f16* wh = (f16*)alloc((size_t)Op * Kw * 2);
      k_wconv<<<blocks((size_t)Op * Kw, 256), 256, 0, stream>>>(fpW[t][j], wh, O, Cw, Op, Kw);
      bool last = (j == FPN[t] - 1);
      f16* dst = (src == ping) ? pong : ping;
      f16* f16o = dst;
      float* f32o = nullptr;
      if (last && lvl > 0) { f32o = fpo[lvl]; f16o = nullptr; }
      dim3 gg((unsigned)(M / 16), (unsigned)(Op / 16));
      k_gemm<<<gg, 32, 0, stream>>>(src, wh, fpB[t][j], f16o, f32o, (int)M, O, Kw, Op, 1);
      src = dst;
    }
    if (lvl > 0) {
      fs[lvl] = fpo[lvl];
      Cf[lvl] = FPSPEC[t][FPN[t] - 1][0];
    } else {
      headsrc = src;
    }
  }

  // --- head: relu(128x128) then 3x128 (no relu) -> d_out (B,N,3) f32
  {
    size_t M = (size_t)B * N;
    {
      int O = HEADSPEC[0][0], Cw = HEADSPEC[0][1];
      int Kw = r32i(Cw), Op = r32i(O);
      f16* wh = (f16*)alloc((size_t)Op * Kw * 2);
      k_wconv<<<blocks((size_t)Op * Kw, 256), 256, 0, stream>>>(hW[0], wh, O, Cw, Op, Kw);
      f16* dst = (headsrc == ping) ? pong : ping;
      dim3 gg((unsigned)(M / 16), (unsigned)(Op / 16));
      k_gemm<<<gg, 32, 0, stream>>>(headsrc, wh, hB[0], dst, (float*)nullptr, (int)M, O, Kw, Op, 1);
      headsrc = dst;
    }
    {
      int O = HEADSPEC[1][0], Cw = HEADSPEC[1][1];  // O=3
      int Kw = r32i(Cw), Op = r32i(O);              // Op=32
      f16* wh = (f16*)alloc((size_t)Op * Kw * 2);
      k_wconv<<<blocks((size_t)Op * Kw, 256), 256, 0, stream>>>(hW[1], wh, O, Cw, Op, Kw);
      dim3 gg((unsigned)(M / 16), (unsigned)(Op / 16));
      k_gemm<<<gg, 32, 0, stream>>>(headsrc, wh, hB[1], (f16*)nullptr, (float*)d_out,
                                    (int)M, O, Kw, Op, 0);
    }
  }
}
